// GraphSAGELayer_49143015800982
// MI455X (gfx1250) — matrix operand order repaired and checked by reading, not compiled, not run on hardware
//
#include <hip/hip_runtime.h>

#define NNODES 50000
#define NEDGES 400000
#define BN_EPS 1e-5f
#define MTILES (NNODES / 16)   // 3125 full 16-row tiles (50000 % 16 == 0)

typedef _Float16 f16;
typedef _Float16 __attribute__((ext_vector_type(8)))  v8h;
typedef _Float16 __attribute__((ext_vector_type(16))) v16h;
typedef float    __attribute__((ext_vector_type(8)))  v8f;

// ---------------------------------------------------------------- utilities
__global__ void zero_f32_kernel(float* __restrict__ p, int n) {
  int i = blockIdx.x * blockDim.x + threadIdx.x;
  if (i < n) p[i] = 0.0f;
}

// ---------------------------------------------------------------- degree
__global__ void deg_scatter_kernel(const int* __restrict__ ei, float* __restrict__ deg) {
  int e = blockIdx.x * blockDim.x + threadIdx.x;
  if (e < NEDGES) unsafeAtomicAdd(&deg[ei[NEDGES + e]], 1.0f);
}

__global__ void deg_invert_kernel(float* __restrict__ deg) {
  int i = blockIdx.x * blockDim.x + threadIdx.x;
  if (i < NNODES) deg[i] = 1.0f / fmaxf(deg[i], 1.0f);
}

// ------------------------------------------------- edge gather + scatter-add
// one thread = one float4 chunk of one edge's feature vector
__global__ void feat_scatter_kernel(const float* __restrict__ h, const int* __restrict__ ei,
                                    float* __restrict__ agg, int di, int cpeShift) {
  int idx = blockIdx.x * blockDim.x + threadIdx.x;
  int e = idx >> cpeShift;
  if (e >= NEDGES) return;
  int f = (idx & ((1 << cpeShift) - 1)) << 2;
  int src = ei[e], dst = ei[NEDGES + e];
  const float4 v = *(const float4*)(h + src * di + f);
  float* a = agg + dst * di + f;
  unsafeAtomicAdd(a + 0, v.x);
  unsafeAtomicAdd(a + 1, v.y);
  unsafeAtomicAdd(a + 2, v.z);
  unsafeAtomicAdd(a + 3, v.w);
}

// ------------------------------------------- build f16 X = [mean_agg | h]  (N x 2di)
__global__ void build_x_kernel(const float* __restrict__ agg, const float* __restrict__ h,
                               const float* __restrict__ deginv, f16* __restrict__ X,
                               int di, int kShift) {
  int i = blockIdx.x * blockDim.x + threadIdx.x;
  int n = i >> kShift;
  if (n >= NNODES) return;
  int k = i & ((1 << kShift) - 1);
  float v = (k < di) ? agg[n * di + k] * deginv[n]
                     : h[n * di + (k - di)];
  X[i] = (f16)v;
}

// ------------------------------------------- build f16 W = [Wl ; Wr]  (2di x do)
__global__ void build_w_kernel(const float* __restrict__ Wl, const float* __restrict__ Wr,
                               f16* __restrict__ Wh, int di, int doShift) {
  int i = blockIdx.x * blockDim.x + threadIdx.x;
  int k = i >> doShift;
  if (k >= 2 * di) return;
  int n = i & ((1 << doShift) - 1);
  float v = (k < di) ? Wl[(k << doShift) + n] : Wr[((k - di) << doShift) + n];
  Wh[i] = (f16)v;
}

// ------------------------------------------------------------ WMMA GEMM
// out[N x Ndim] = X[N x K] @ W[K x Ndim] + bias, f16 operands, f32 accum.
// block = 256 threads = 8 waves; wave computes 64x16; block computes 512x16.
// Full B panel (K x 16, <= 32 KB) staged to LDS once, in per-lane fragment layout.
// k-loop software-pipelined by one step: next A/B fragments prefetched before
// issuing the current step's WMMAs.
__device__ __forceinline__ v16h load_a_frag(const f16* __restrict__ p) {
  v8h lo = *(const v8h*)p;          // K = kbase + 0..7
  v8h hi = *(const v8h*)(p + 16);   // K = kbase + 16..23
  return __builtin_shufflevector(lo, hi, 0,1,2,3,4,5,6,7,8,9,10,11,12,13,14,15);
}

#define WMMA_F16(A, B, C) \
  __builtin_amdgcn_wmma_f32_16x16x32_f16(false, (A), false, (B), (short)0, (C), false, false)

__global__ __launch_bounds__(256)
void sage_gemm_wmma_kernel(const f16* __restrict__ X, const f16* __restrict__ W,
                           const float* __restrict__ bias, float* __restrict__ out,
                           int K, int Ndim) {
  __shared__ __align__(32) f16 ldsB[16384];   // up to K=1024: 1024x16 halves = 32 KB
  const int tid   = threadIdx.x;
  const int wave  = tid >> 5;
  const int lane  = tid & 31;
  const int lhalf = lane & 15;
  const int lhi   = lane >> 4;
  const int colBase  = blockIdx.x << 4;
  const int tileBase = blockIdx.y * 32 + (wave << 2);

  // ---- stage whole B panel in fragment layout:
  // k-step kb (32 Ks) lives at ldsB[kb*512 + lane*16 + j],
  // lane = col + 16*((ks>>3)&1), element j = (ks & 7) + 8*(ks>=16) (ks = K within step)
  for (int i = tid; i < (K << 4); i += 256) {
    int k = i >> 4, n = i & 15;             // 16 consecutive threads: 32B row read
    int ks = k & 31, kb = k >> 5;
    ldsB[(kb << 9) + ((n + ((ks & 8) << 1)) << 4) + ((ks & 7) | ((ks & 16) >> 1))] = W[k * Ndim + colBase + n];
  }
  __syncthreads();

  // ---- clamp tiles (redundant compute for tail waves; stores guarded later)
  const int tLast = MTILES - 1;
  int t0 = min(tileBase + 0, tLast);
  int t1 = min(tileBase + 1, tLast);
  int t2 = min(tileBase + 2, tLast);
  int t3 = min(tileBase + 3, tLast);
  const f16* a0 = X + (t0 * 16 + lhalf) * K + (lhi << 3);
  const f16* a1 = X + (t1 * 16 + lhalf) * K + (lhi << 3);
  const f16* a2 = X + (t2 * 16 + lhalf) * K + (lhi << 3);
  const f16* a3 = X + (t3 * 16 + lhalf) * K + (lhi << 3);

  v8f acc0 = {}, acc1 = {}, acc2 = {}, acc3 = {};

  // ---- prologue: fragments for k0 = 0
  v16h b  = *(const v16h*)&ldsB[lane << 4];
  v16h f0 = load_a_frag(a0);
  v16h f1 = load_a_frag(a1);
  v16h f2 = load_a_frag(a2);
  v16h f3 = load_a_frag(a3);

  const int kLast = K - 32;
#pragma unroll 2
  for (int k0 = 0; k0 < kLast; k0 += 32) {
    const int kn = k0 + 32;
    // prefetch next step's fragments before consuming current ones
    v16h bn = *(const v16h*)&ldsB[(kn << 4) + (lane << 4)];
    v16h g0 = load_a_frag(a0 + kn);
    v16h g1 = load_a_frag(a1 + kn);
    v16h g2 = load_a_frag(a2 + kn);
    v16h g3 = load_a_frag(a3 + kn);
    acc0 = WMMA_F16(f0, b, acc0);
    acc1 = WMMA_F16(f1, b, acc1);
    acc2 = WMMA_F16(f2, b, acc2);
    acc3 = WMMA_F16(f3, b, acc3);
    b = bn; f0 = g0; f1 = g1; f2 = g2; f3 = g3;
  }
  // ---- epilogue k-step
  acc0 = WMMA_F16(f0, b, acc0);
  acc1 = WMMA_F16(f1, b, acc1);
  acc2 = WMMA_F16(f2, b, acc2);
  acc3 = WMMA_F16(f3, b, acc3);

  // ---- bias + store (wave-uniform guards, outside hot loop)
  const int   col  = colBase + lhalf;
  const float bcol = bias[col];
  const int rb = lhi << 3;   // C layout: VGPR r -> M = r + 8*(lane>=16)
  if (tileBase + 0 < MTILES) {
    float* o = out + ((tileBase + 0) * 16 + rb) * Ndim + col;
#pragma unroll
    for (int r = 0; r < 8; ++r) o[r * Ndim] = acc0[r] + bcol;
  }
  if (tileBase + 1 < MTILES) {
    float* o = out + ((tileBase + 1) * 16 + rb) * Ndim + col;
#pragma unroll
    for (int r = 0; r < 8; ++r) o[r * Ndim] = acc1[r] + bcol;
  }
  if (tileBase + 2 < MTILES) {
    float* o = out + ((tileBase + 2) * 16 + rb) * Ndim + col;
#pragma unroll
    for (int r = 0; r < 8; ++r) o[r * Ndim] = acc2[r] + bcol;
  }
  if (tileBase + 3 < MTILES) {
    float* o = out + ((tileBase + 3) * 16 + rb) * Ndim + col;
#pragma unroll
    for (int r = 0; r < 8; ++r) o[r * Ndim] = acc3[r] + bcol;
  }
}

// ------------------------------------------------------------ BatchNorm
// coalesced partial sums: block handles 512 rows, thread owns column(s) tid (+256)
__global__ void bn_partial_kernel(const float* __restrict__ out, float* __restrict__ stats,
                                  int Ndim) {
  int r0   = blockIdx.x * 512;
  int rend = min(r0 + 512, NNODES);
  int j0   = threadIdx.x;
  bool two = (Ndim > 256);
  float s0 = 0.f, q0 = 0.f, s1 = 0.f, q1 = 0.f;
  for (int r = r0; r < rend; ++r) {
    const float* row = out + r * Ndim;
    float v = row[j0]; s0 += v; q0 += v * v;
    if (two) { float w = row[j0 + 256]; s1 += w; q1 += w * w; }
  }
  unsafeAtomicAdd(&stats[j0], s0);
  unsafeAtomicAdd(&stats[512 + j0], q0);
  if (two) {
    unsafeAtomicAdd(&stats[j0 + 256], s1);
    unsafeAtomicAdd(&stats[512 + j0 + 256], q1);
  }
}

__global__ void bn_finalize_kernel(float* __restrict__ stats, int Ndim) {
  int j = blockIdx.x * blockDim.x + threadIdx.x;
  if (j >= Ndim) return;
  const float invN = 1.0f / (float)NNODES;
  float mu  = stats[j] * invN;
  float var = stats[512 + j] * invN - mu * mu;   // biased variance (matches jnp var)
  stats[j]       = mu;
  stats[512 + j] = rsqrtf(fmaxf(var, 0.0f) + BN_EPS);
}

__global__ void bn_apply_kernel(float* __restrict__ out, const float* __restrict__ stats,
                                const float* __restrict__ gamma, const float* __restrict__ beta,
                                int doShift) {
  int i = blockIdx.x * blockDim.x + threadIdx.x;
  int n = i >> doShift;
  if (n >= NNODES) return;
  int j = i & ((1 << doShift) - 1);
  float v = out[i];
  v = (v - stats[j]) * stats[512 + j] * gamma[j] + beta[j];
  out[i] = fmaxf(v, 0.0f);
}

// ---------------------------------------------------------------- launcher
extern "C" void kernel_launch(void* const* d_in, const int* in_sizes, int n_in,
                              void* d_out, int out_size, void* d_ws, size_t ws_size,
                              hipStream_t stream) {
  (void)in_sizes; (void)n_in; (void)out_size; (void)ws_size;

  const float* x  = (const float*)d_in[0];
  const int*   ei = (const int*)d_in[1];

  const float* Wl[3] = {(const float*)d_in[2], (const float*)d_in[7],  (const float*)d_in[12]};
  const float* bl[3] = {(const float*)d_in[3], (const float*)d_in[8],  (const float*)d_in[13]};
  const float* Wr[3] = {(const float*)d_in[4], (const float*)d_in[9],  (const float*)d_in[14]};
  const float* gm[3] = {(const float*)d_in[5], (const float*)d_in[10], (const float*)d_in[15]};
  const float* bt[3] = {(const float*)d_in[6], (const float*)d_in[11], (const float*)d_in[16]};
  const int di[3]      = {256, 512, 512};
  const int dO[3]      = {512, 512, 256};
  const int doShift[3] = {9, 9, 8};

  char* ws = (char*)d_ws;
  size_t off = 0;
  auto carve = [&](size_t bytes) -> char* {
    char* p = ws + off;
    off = (off + bytes + 255) & ~(size_t)255;
    return p;
  };
  float* deg   = (float*)carve((size_t)NNODES * 4);
  float* agg   = (float*)carve((size_t)NNODES * 512 * 4);
  float* hA    = (float*)carve((size_t)NNODES * 512 * 4);
  float* hB    = (float*)carve((size_t)NNODES * 512 * 4);
  f16*   Xh    = (f16*)  carve((size_t)NNODES * 1024 * 2);
  f16*   Wh    = (f16*)  carve((size_t)1024 * 512 * 2);
  float* stats = (float*)carve(1024 * 4);

  const int T = 256;
  // 1 / max(deg, 1), shared by all layers
  zero_f32_kernel<<<(NNODES + T - 1) / T, T, 0, stream>>>(deg, NNODES);
  deg_scatter_kernel<<<(NEDGES + T - 1) / T, T, 0, stream>>>(ei, deg);
  deg_invert_kernel<<<(NNODES + T - 1) / T, T, 0, stream>>>(deg);

  const float* hin = x;
  float* hout[3] = {hA, hB, (float*)d_out};

  for (int L = 0; L < 3; ++L) {
    const int DI = di[L], DO = dO[L], K = 2 * DI;
    const int cpeShift = (DI == 256) ? 6 : 7;   // log2(DI/4)
    const int kShift   = (K == 512) ? 9 : 10;   // log2(2*DI)

    const int nAgg = NNODES * DI;
    zero_f32_kernel<<<(nAgg + T - 1) / T, T, 0, stream>>>(agg, nAgg);

    const int nScat = NEDGES << cpeShift;
    feat_scatter_kernel<<<(nScat + T - 1) / T, T, 0, stream>>>(hin, ei, agg, DI, cpeShift);

    const int nX = NNODES * K;
    build_x_kernel<<<(nX + T - 1) / T, T, 0, stream>>>(agg, hin, deg, Xh, DI, kShift);

    const int nW = K * DO;
    build_w_kernel<<<(nW + T - 1) / T, T, 0, stream>>>(Wl[L], Wr[L], Wh, DI, doShift[L]);

    dim3 gg(DO / 16, (MTILES + 31) / 32);       // (32|16, 98)
    sage_gemm_wmma_kernel<<<gg, 256, 0, stream>>>(Xh, Wh, bl[L], hout[L], K, DO);

    zero_f32_kernel<<<(1024 + T - 1) / T, T, 0, stream>>>(stats, 1024);
    bn_partial_kernel<<<(NNODES + 511) / 512, 256, 0, stream>>>(hout[L], stats, DO);
    bn_finalize_kernel<<<(DO + T - 1) / T, T, 0, stream>>>(stats, DO);

    const int nOut = NNODES * DO;
    bn_apply_kernel<<<(nOut + T - 1) / T, T, 0, stream>>>(hout[L], stats, gm[L], bt[L], doShift[L]);

    hin = hout[L];
  }
}
